// PointCloud3DCNN_62268435857938
// MI455X (gfx1250) — matrix operand order repaired and checked by reading, not compiled, not run on hardware
//
#include <hip/hip_runtime.h>

#define DEV_INLINE __device__ __forceinline__

typedef __attribute__((ext_vector_type(16))) __bf16 v16bf;
typedef __attribute__((ext_vector_type(8)))  float  v8f;
typedef __attribute__((__vector_size__(16))) int vi4;

union BVec { v16bf v; uint4 q[2]; };

#if defined(__has_builtin)
#if __has_builtin(__builtin_amdgcn_global_load_async_to_lds_b128) && \
    __has_builtin(__builtin_amdgcn_s_wait_asynccnt)
#define ASYNC_LDS 1
#endif
#endif
#ifndef ASYNC_LDS
#define ASYNC_LDS 0
#endif

DEV_INLINE void async_copy_b128(const unsigned short* g, unsigned short* l) {
#if ASYNC_LDS
  __builtin_amdgcn_global_load_async_to_lds_b128(
      (__attribute__((address_space(1))) vi4*)(unsigned long long)g,
      (__attribute__((address_space(3))) vi4*)(unsigned int)(unsigned long long)l, 0, 0);
#else
  *(uint4*)l = *(const uint4*)g;
#endif
}

template <int N>
DEV_INLINE void wait_async() {
#if ASYNC_LDS
  __builtin_amdgcn_s_wait_asynccnt(N);
#endif
}

DEV_INLINE unsigned short f32_to_bf16(float f) {
  unsigned int u = __float_as_uint(f);
  u += 0x7fffu + ((u >> 16) & 1u);          // round-to-nearest-even
  return (unsigned short)(u >> 16);
}
DEV_INLINE float bf16_to_f32(unsigned short h) {
  return __uint_as_float(((unsigned int)h) << 16);
}

// packed voxel code: (b<<18)|(z<<12)|(y<<6)|x   (all coords < 64, b < 2)
DEV_INLINE int pack_code(int b, int z, int y, int x) {
  return (b << 18) | (z << 12) | (y << 6) | x;
}

// ---------------------------------------------------------------- utilities
__global__ void fill_zero_kernel(unsigned int* __restrict__ p, long long n) {
  long long t = (long long)blockIdx.x * blockDim.x + threadIdx.x;
  if (t < n) p[t] = 0u;
}

// scatter level-0 features into channel-padded (stride 16) bf16 grid
__global__ void scatter_l0_kernel(const float* __restrict__ vf, const int* __restrict__ idx,
                                  unsigned short* __restrict__ grid, unsigned char* __restrict__ mask,
                                  int* __restrict__ list, int total, int S, int Cdata, int Cpad,
                                  int P_) {
  int t = blockIdx.x * blockDim.x + threadIdx.x;
  if (t >= total) return;
  int b = t / P_;
  const int* q = idx + (size_t)t * 3;
  size_t base = (((size_t)b * S + q[0]) * S + q[1]) * S + q[2];
  mask[base] = 1;
  list[t] = pack_code(b, q[0], q[1], q[2]);
  const float* f = vf + (size_t)t * Cdata;
  for (int c = 0; c < Cdata; ++c) grid[base * Cpad + c] = f32_to_bf16(f[c]);
}

// active-output set of a k3 s2 p0 sparse conv: OR over the 3x3x3 input window
__global__ void downsample_mask_kernel(const unsigned char* __restrict__ mIn, int Sin,
                                       unsigned char* __restrict__ mOut, int* __restrict__ list,
                                       int* __restrict__ counter, int Sout, int total) {
  int t = blockIdx.x * blockDim.x + threadIdx.x;
  if (t >= total) return;
  int x = t % Sout, r = t / Sout;
  int y = r % Sout; r /= Sout;
  int z = r % Sout; int b = r / Sout;
  bool any = false;
  for (int kd = 0; kd < 3; ++kd)
    for (int kh = 0; kh < 3; ++kh)
      for (int kw = 0; kw < 3; ++kw) {
        int zz = 2 * z + kd, yy = 2 * y + kh, xx = 2 * x + kw;
        any = any || (mIn[((size_t)(b * Sin + zz) * Sin + yy) * Sin + xx] != 0);
      }
  if (any) {
    mOut[((size_t)(b * Sout + z) * Sout + y) * Sout + x] = 1;
    int pos = atomicAdd(counter, 1);
    list[pos] = pack_code(b, z, y, x);
  }
}

// Repack OIDHW f32 weights -> fragment-ordered bf16 B-tiles.
// K ordering: K = n*CinPad + ci with n = (kd*3+kh)*3+kw; ci >= CinW contributes zero.
__global__ void pack_weights_kernel(const float* __restrict__ w, int CinPad, int CinW, int Cout,
                                    unsigned short* __restrict__ out, int total) {
  int t = blockIdx.x * blockDim.x + threadIdx.x;
  if (t >= total) return;
  int e = t & 15;
  int lane = (t >> 4) & 31;
  int rest = t >> 9;
  int nTilesN = Cout >> 4;
  int tIdx = rest % nTilesN;
  int chunk = rest / nTilesN;
  int N = tIdx * 16 + (lane & 15);
  int hf = lane >> 4;
  int K = chunk * 32 + ((e & 8) << 1) + hf * 8 + (e & 7);
  float v = 0.0f;
  int Ktot = 27 * CinPad;
  if (K < Ktot) {
    int n = K / CinPad, ci = K - n * CinPad;
    if (ci < CinW) v = w[((size_t)N * CinW + ci) * 27 + n];
  }
  out[t] = f32_to_bf16(v);
}

// fold conv bias + eval-mode BN into per-channel alpha/beta: y = dot*alpha + beta
__global__ void fold_bn_kernel(const float* __restrict__ b, const float* __restrict__ s,
                               const float* __restrict__ sb, float* __restrict__ alpha,
                               float* __restrict__ beta, int C) {
  int t = blockIdx.x * blockDim.x + threadIdx.x;
  if (t >= C) return;
  alpha[t] = s[t];
  beta[t] = b[t] * s[t] + sb[t];
}

// ---------------------------------------------------------------- WMMA conv
// MODE 0: submanifold k3 s1 p1;  MODE 1: down k3 s2 p0;  MODE 2: inverse k3 s2 (lhs-dilated)
// CIN: power-of-two channel stride of the input grid.
// Block = 1 M-tile x NW N-tiles (one wave per N-tile). Per K-chunk the 16x32 A panel is
// staged once into LDS via per-lane async gather (triple-buffered), shared by all waves.
// ISSUES = async instructions per stage per issuing wave (immediate for s_wait_asynccnt).
template <int CIN, int MODE, int ISSUES>
__global__ void conv_wmma_kernel(const unsigned short* __restrict__ gin, int Sin,
                                 unsigned short* __restrict__ gout, int Sout, int Cout,
                                 const int* __restrict__ list, const int* __restrict__ cntPtr,
                                 int fixedCnt,
                                 const float* __restrict__ alpha, const float* __restrict__ beta,
                                 const unsigned short* __restrict__ wpk,
                                 const unsigned short* __restrict__ resid,
                                 const unsigned short* __restrict__ zpad) {
  static_assert((CIN & 7) == 0, "CIN must be a multiple of 8 for b128 A-gather");
  const int tid = threadIdx.x;
  const int lane = tid & 31;
  const int wave = tid >> 5;
  const int NW = blockDim.x >> 5;
  const int tile = blockIdx.x;
  const int nTile = blockIdx.y * NW + wave;   // exact: nTilesN % NW == 0 by construction
  const int nTilesN = Cout >> 4;
  const int cnt = cntPtr ? cntPtr[0] : fixedCnt;
  if (tile * 16 >= cnt) return;

  __shared__ __align__(16) unsigned short ldsA[3 * 512];  // triple-buffered 16x32 bf16 panel
  __shared__ int sB[16], sZ[16], sY[16], sX[16];
  if (tid < 16) {
    int v = tile * 16 + tid;
    int code = list[v < cnt ? v : tile * 16];
    sX[tid] = code & 63;
    sY[tid] = (code >> 6) & 63;
    sZ[tid] = (code >> 12) & 63;
    sB[tid] = code >> 18;
  }
  __syncthreads();

  const int Ktot = 27 * CIN;
  const int Kpad = (Ktot + 31) & ~31;
  const int nChunks = Kpad / 32;

  // stage one K-chunk's A panel: 64 runs of 8 bf16 -> one b128 async gather each
  auto stage = [&](int chunk) {
    const int kb = chunk * 32;
    unsigned short* buf = ldsA + (chunk % 3) * 512;
    for (int r = tid; r < 64; r += blockDim.x) {
      const int Mr = r & 15;
      const int runI = r >> 4;                 // K offset = runI*8
      const int K0 = kb + runI * 8;
      const unsigned short* src = zpad;        // zero page for padded/invalid runs
      if (((tile * 16 + Mr) < cnt) && (K0 < Ktot)) {
        const int n = K0 / CIN;                // shift (CIN power of two)
        const int ci = K0 - n * CIN;
        const int kd = n / 9, kh = (n / 3) % 3, kw = n % 3;
        const int vz = sZ[Mr], vy = sY[Mr], vx = sX[Mr];
        int zz, yy, xx;
        bool ok = true;
        if (MODE == 0) {
          zz = vz + kd - 1; yy = vy + kh - 1; xx = vx + kw - 1;
        } else if (MODE == 1) {
          zz = 2 * vz + kd; yy = 2 * vy + kh; xx = 2 * vx + kw;
        } else {
          int jz = vz + kd - 2, jy = vy + kh - 2, jx = vx + kw - 2;
          ok = ((jz | jy | jx) & 1) == 0;      // all dilated indices even
          zz = jz >> 1; yy = jy >> 1; xx = jx >> 1;
        }
        ok = ok && zz >= 0 && zz < Sin && yy >= 0 && yy < Sin && xx >= 0 && xx < Sin;
        if (ok)
          src = gin + ((((size_t)sB[Mr] * Sin + zz) * Sin + yy) * Sin + xx) * CIN + ci;
      }
      async_copy_b128(src, buf + (size_t)r * 8);
    }
  };

  const int M = lane & 15;
  const int hf = lane >> 4;
  v8f acc = {};

  stage(0);
  for (int c = 0; c < nChunks; ++c) {
    if (c + 1 < nChunks) {
      stage(c + 1);
      wait_async<ISSUES>();   // chunk c landed; chunk c+1 may still be in flight
    } else {
      wait_async<0>();
    }
    __syncthreads();          // triple buffer => one barrier per chunk suffices

    const unsigned short* buf = ldsA + (c % 3) * 512;
    BVec av;                  // lane reads runs hf and hf+2 (K = kb+hf*8, kb+16+hf*8)
    av.q[0] = *(const uint4*)(buf + ((hf * 16 + M) * 8));
    av.q[1] = *(const uint4*)(buf + (((hf + 2) * 16 + M) * 8));

    const uint4* wq = (const uint4*)(wpk + (((size_t)c * nTilesN + nTile) * 32 + lane) * 16);
    __builtin_prefetch((const char*)wq + (size_t)nTilesN * 1024, 0, 1);
    BVec bv;
    bv.q[0] = wq[0];
    bv.q[1] = wq[1];
    acc = __builtin_amdgcn_wmma_f32_16x16x32_bf16(false, av.v, false, bv.v,
                                                  (short)0, acc, false, false);
  }

  // ---- epilogue: alpha/beta (bias+BN), ReLU, optional residual, bf16 scatter ----
  const int N = nTile * 16 + M;
  const float al = alpha[N];
  const float be = beta[N];
#pragma unroll
  for (int r = 0; r < 8; ++r) {
    const int Mr = r + 8 * hf;
    const int vox = tile * 16 + Mr;
    if (vox < cnt) {
      const size_t addr =
          ((((size_t)sB[Mr] * Sout + sZ[Mr]) * Sout + sY[Mr]) * Sout + sX[Mr]) * Cout + N;
      float val = fmaxf(acc[r] * al + be, 0.0f);
      if (resid) val += bf16_to_f32(resid[addr]);
      gout[addr] = f32_to_bf16(val);
    }
  }
}

// ---------------------------------------------------------------- heads
__global__ void occu_kernel(const unsigned short* __restrict__ dec0, const float* __restrict__ cw,
                            const float* __restrict__ cb, float* __restrict__ out, int S, int C,
                            int total) {
  int t = blockIdx.x * blockDim.x + threadIdx.x;
  if (t >= total) return;
  int x = t % S, r = t / S;
  int y = r % S; r /= S;
  int z = r % S; int b = r / S;
  float acc = cb[0];
  for (int kd = 0; kd < 3; ++kd) {
    int zz = z + kd - 1; if (zz < 0 || zz >= S) continue;
    for (int kh = 0; kh < 3; ++kh) {
      int yy = y + kh - 1; if (yy < 0 || yy >= S) continue;
      for (int kw = 0; kw < 3; ++kw) {
        int xx = x + kw - 1; if (xx < 0 || xx >= S) continue;
        const unsigned short* p = dec0 + ((((size_t)b * S + zz) * S + yy) * S + xx) * C;
        int n = (kd * 3 + kh) * 3 + kw;
        for (int c = 0; c < C; ++c) acc += bf16_to_f32(p[c]) * cw[c * 27 + n];
      }
    }
  }
  out[t] = acc;
}

__global__ void fc_kernel(const unsigned short* __restrict__ dec0, const int* __restrict__ idx,
                          const float* __restrict__ w, const float* __restrict__ bia,
                          float* __restrict__ out, int total, int P_, int S, int C) {
  int t = blockIdx.x * blockDim.x + threadIdx.x;
  if (t >= total) return;
  int b = t / P_;
  const int* q = idx + (size_t)t * 3;
  size_t base = ((((size_t)b * S + q[0]) * S + q[1]) * S + q[2]) * C;
  float f[16];
  for (int c = 0; c < 16; ++c) f[c] = bf16_to_f32(dec0[base + c]);
  for (int k = 0; k < 3; ++k) {
    float a = bia[k];
    for (int c = 0; c < 16; ++c) a += f[c] * w[k * 16 + c];
    out[(size_t)t * 3 + k] = a;
  }
}

// ---------------------------------------------------------------- dispatch
// ISSUES is static per case: Cout==16 layers run as a single-wave block (2 async
// issues per stage); all others use >=2 waves (1 issue per stage).
static inline void launch_conv(int Cin, int mode, int maxTiles, hipStream_t stream,
                               const unsigned short* gin, int Sin, unsigned short* gout, int Sout,
                               int Cout, const int* list, const int* cntPtr, int fixedCnt,
                               const float* alpha, const float* beta, const unsigned short* wpk,
                               const unsigned short* resid, const unsigned short* zpad) {
  const int nTilesN = Cout >> 4;
  const int nw = (nTilesN < 4) ? nTilesN : 4;  // nTilesN in {1,2,4,8,16} => divides exactly
  dim3 g(maxTiles, nTilesN / nw);
  dim3 blk(32 * nw);
#define CW_CASE(CI, MO, IS)                                                                   \
  if (Cin == CI && mode == MO) {                                                              \
    conv_wmma_kernel<CI, MO, IS><<<g, blk, 0, stream>>>(gin, Sin, gout, Sout, Cout, list,     \
                                                        cntPtr, fixedCnt, alpha, beta, wpk,   \
                                                        resid, zpad);                         \
    return;                                                                                   \
  }
  // SubM:  (16,0) always Cout=16 (enc1, dec0); larger CIN always Cout=CIN
  CW_CASE(16, 0, 2) CW_CASE(32, 0, 1) CW_CASE(64, 0, 1) CW_CASE(128, 0, 1) CW_CASE(256, 0, 1)
  // Down:  Cout = 2*CIN >= 32
  CW_CASE(16, 1, 1) CW_CASE(32, 1, 1) CW_CASE(64, 1, 1) CW_CASE(128, 1, 1)
  // Up:    Cout = CIN/2; (32,2) -> Cout=16 single-wave
  CW_CASE(32, 2, 2) CW_CASE(64, 2, 1) CW_CASE(128, 2, 1) CW_CASE(256, 2, 1)
#undef CW_CASE
}

extern "C" void kernel_launch(void* const* d_in, const int* in_sizes, int n_in, void* d_out,
                              int out_size, void* d_ws, size_t ws_size, hipStream_t stream) {
  (void)in_sizes; (void)n_in; (void)out_size; (void)ws_size;
  const int B = 2, P = 5000;
  const int S[5] = {63, 31, 15, 7, 3};
  const int C[5] = {16, 32, 64, 128, 256};
  int V[5];
  for (int l = 0; l < 5; ++l) V[l] = S[l] * S[l] * S[l];

  // params flattened in jax pytree order (top keys sorted):
  // conv_b(2), conv_w(3), down[0..3]x{b,b2,s,s2,sb,sb2,w,w2}(4..35),
  // enc1{b,s,sb,w}(36..39), fc1_b(40), fc1_w(41), up[0..3]x{...}(42..73)
  const float* vox_f = (const float*)d_in[0];
  const int* idx = (const int*)d_in[1];
  const float* conv_b = (const float*)d_in[2];
  const float* conv_w = (const float*)d_in[3];
  const int DOWN0 = 4, ENC1 = 36, FC1B = 40, FC1W = 41, UP0 = 42;

  size_t off = 0;
  auto alloc = [&](size_t bytes) -> char* {
    off = (off + 255) & ~(size_t)255;
    char* p = (char*)d_ws + off;
    off += bytes;
    return p;
  };
  auto zero = [&](void* p, size_t bytes) {
    long long n = (long long)((bytes + 3) / 4);
    long long blk = (n + 255) / 256;
    fill_zero_kernel<<<dim3((unsigned)blk), 256, 0, stream>>>((unsigned int*)p, n);
  };

  int* counters = (int*)alloc(8 * sizeof(int));
  unsigned short* zpad = (unsigned short*)alloc(64);                        // zero page
  unsigned short* in0 = (unsigned short*)alloc((size_t)B * V[0] * 16 * 2);  // 9ch padded to 16
  unsigned short *encG[5], *tmpG[5], *decG[4];
  unsigned char* maskG[5];
  int* listG[5];
  for (int l = 0; l < 5; ++l) encG[l] = (unsigned short*)alloc((size_t)B * V[l] * C[l] * 2);
  for (int l = 0; l < 5; ++l) tmpG[l] = (unsigned short*)alloc((size_t)B * V[l] * C[l] * 2);
  for (int l = 0; l < 4; ++l) decG[l] = (unsigned short*)alloc((size_t)B * V[l] * C[l] * 2);
  for (int l = 0; l < 5; ++l) maskG[l] = (unsigned char*)alloc((size_t)B * V[l]);
  for (int l = 0; l < 5; ++l) listG[l] = (int*)alloc((size_t)B * V[l] * sizeof(int));

  zero(counters, 8 * sizeof(int));
  zero(zpad, 64);
  zero(in0, (size_t)B * V[0] * 16 * 2);
  for (int l = 0; l < 5; ++l) {
    zero(encG[l], (size_t)B * V[l] * C[l] * 2);
    zero(tmpG[l], (size_t)B * V[l] * C[l] * 2);
    zero(maskG[l], (size_t)B * V[l]);
  }
  for (int l = 0; l < 4; ++l) zero(decG[l], (size_t)B * V[l] * C[l] * 2);

  {  // scatter active voxels into the level-0 grid + build mask/list
    int total = B * P;
    scatter_l0_kernel<<<(total + 255) / 256, 256, 0, stream>>>(vox_f, idx, in0, maskG[0],
                                                               listG[0], total, S[0], 9, 16, P);
  }
  for (int l = 0; l < 4; ++l) {  // active sets at coarser levels
    int total = B * V[l + 1];
    downsample_mask_kernel<<<(total + 255) / 256, 256, 0, stream>>>(
        maskG[l], S[l], maskG[l + 1], listG[l + 1], counters + (l + 1), S[l + 1], total);
  }

  auto run_conv = [&](const unsigned short* gin, int Sin, int CinPad, int CinW,
                      unsigned short* gout, int Sout, int Cout, int lvlOut, int w_i, int b_i,
                      int s_i, int sb_i, const unsigned short* resid, int mode) {
    const int Ktot = 27 * CinPad, Kpad = (Ktot + 31) & ~31, nT = Cout >> 4;
    unsigned short* wpk = (unsigned short*)alloc((size_t)Kpad * Cout * 2);
    float* alpha = (float*)alloc((size_t)Cout * 4);
    float* beta = (float*)alloc((size_t)Cout * 4);
    int totalPack = (Kpad / 32) * nT * 512;
    pack_weights_kernel<<<(totalPack + 255) / 256, 256, 0, stream>>>(
        (const float*)d_in[w_i], CinPad, CinW, Cout, wpk, totalPack);
    fold_bn_kernel<<<(Cout + 63) / 64, 64, 0, stream>>>(
        (const float*)d_in[b_i], (const float*)d_in[s_i], (const float*)d_in[sb_i], alpha, beta,
        Cout);
    const int maxTiles = (lvlOut == 0) ? (B * P + 15) / 16 : (B * V[lvlOut] + 15) / 16;
    const int* cp = (lvlOut == 0) ? nullptr : (counters + lvlOut);
    launch_conv(CinPad, mode, maxTiles, stream, gin, Sin, gout, Sout, Cout, listG[lvlOut], cp,
                B * P, alpha, beta, wpk, resid, zpad);
  };

  // Encoder1 (SubM 9->16, input padded to 16ch)   enc1 keys sorted: b,s,sb,w
  run_conv(in0, S[0], 16, 9, encG[0], S[0], 16, 0, ENC1 + 3, ENC1 + 0, ENC1 + 1, ENC1 + 2,
           nullptr, 0);

  // Encoder2..5: down (k3 s2 p0) then SubM.  down keys sorted: b,b2,s,s2,sb,sb2,w,w2
  for (int L = 0; L < 4; ++L) {
    int base = DOWN0 + L * 8;
    run_conv(encG[L], S[L], C[L], C[L], tmpG[L + 1], S[L + 1], C[L + 1], L + 1, base + 6,
             base + 0, base + 2, base + 4, nullptr, 1);
    run_conv(tmpG[L + 1], S[L + 1], C[L + 1], C[L + 1], encG[L + 1], S[L + 1], C[L + 1], L + 1,
             base + 7, base + 1, base + 3, base + 5, nullptr, 0);
  }

  // Decoder5..2: inverse conv then SubM (+ residual for lvl>0)
  const unsigned short* y = encG[4];
  for (int i = 0; i < 4; ++i) {
    int lvl = 3 - i;
    int base = UP0 + i * 8;
    run_conv(y, S[lvl + 1], C[lvl + 1], C[lvl + 1], tmpG[lvl], S[lvl], C[lvl], lvl, base + 6,
             base + 0, base + 2, base + 4, nullptr, 2);
    run_conv(tmpG[lvl], S[lvl], C[lvl], C[lvl], decG[lvl], S[lvl], C[lvl], lvl, base + 7,
             base + 1, base + 3, base + 5, (lvl > 0) ? encG[lvl] : nullptr, 0);
    y = decG[lvl];
  }

  float* out = (float*)d_out;
  {  // dense occupancy head: Conv3d(16,1,3,pad=1) over the full grid
    int total = B * V[0];
    occu_kernel<<<(total + 255) / 256, 256, 0, stream>>>(decG[0], conv_w, conv_b,
                                                         out + (size_t)B * P * 3, S[0], 16, total);
  }
  {  // coords head: gather + fc1 (16->3)
    int total = B * P;
    fc_kernel<<<(total + 255) / 256, 256, 0, stream>>>(decG[0], idx, (const float*)d_in[FC1W],
                                                       (const float*)d_in[FC1B], out, total, P,
                                                       S[0], 16);
  }
}
